// SimCLRPointCloud_47528108098109
// MI455X (gfx1250) — matrix operand order repaired and checked by reading, compiled, not run on hardware
//
#include <hip/hip_runtime.h>

#define BN   32
#define PP   1024
#define KNN  20
#define NTOT (BN * PP)

typedef __attribute__((ext_vector_type(16))) _Float16 v16h;
typedef __attribute__((ext_vector_type(8)))  _Float16 v8h;
typedef __attribute__((ext_vector_type(8)))  float    v8f;

// ---------------------------------------------------------------------------
// register-resident top-K (K=20) insertion
// ---------------------------------------------------------------------------
__device__ __forceinline__ void topk_insert(float (&bd)[KNN], int (&bi)[KNN],
                                            float d, int j) {
  if (d >= bd[KNN - 1]) return;
  float cd = d; int cj = j;
#pragma unroll
  for (int s = 0; s < KNN; ++s) {
    bool sw = cd < bd[s];
    float td = bd[s]; int tj = bi[s];
    if (sw) { bd[s] = cd; bi[s] = cj; cd = td; cj = tj; }
  }
}

// ---------------------------------------------------------------------------
// Kernel 1: kNN in 3-D position space (per cloud). grid=128, block=256.
// ---------------------------------------------------------------------------
__global__ void knn3_kernel(const float* __restrict__ pos, int* __restrict__ idx1) {
  __shared__ float tile[256 * 3];
  const int blk = blockIdx.x;
  const int b   = blk >> 2;
  const int i   = b * PP + (blk & 3) * 256 + threadIdx.x;
  const float xi0 = pos[i * 3 + 0], xi1 = pos[i * 3 + 1], xi2 = pos[i * 3 + 2];
  float bd[KNN]; int bi[KNN];
#pragma unroll
  for (int s = 0; s < KNN; ++s) { bd[s] = 1e30f; bi[s] = 0; }
  for (int t = 0; t < 4; ++t) {
    const int j0 = b * PP + t * 256;
    for (int u = threadIdx.x; u < 768; u += 256) tile[u] = pos[j0 * 3 + u];
    __syncthreads();
    for (int jj = 0; jj < 256; ++jj) {
      float d0 = xi0 - tile[jj * 3 + 0];
      float d1 = xi1 - tile[jj * 3 + 1];
      float d2 = xi2 - tile[jj * 3 + 2];
      topk_insert(bd, bi, d0 * d0 + d1 * d1 + d2 * d2, j0 + jj);
    }
    __syncthreads();
  }
#pragma unroll
  for (int s = 0; s < KNN; ++s) idx1[i * KNN + s] = bi[s];
}

// ---------------------------------------------------------------------------
// Kernel 2: EdgeConv1 (6->64 BN ReLU, 64->64, max over K). grid=8192, block=256
// (4 points per block, 64 threads/point = one output channel each).
// ---------------------------------------------------------------------------
__global__ void edgeconv1_kernel(const float* __restrict__ pos,
                                 const int*   __restrict__ idx1,
                                 const float* __restrict__ W1, const float* __restrict__ b1,
                                 const float* __restrict__ bns, const float* __restrict__ bnb,
                                 const float* __restrict__ W2, const float* __restrict__ b2,
                                 float* __restrict__ x1f, _Float16* __restrict__ xcat) {
  __shared__ float W2l[64 * 64];
  __shared__ float W1l[6 * 64];
  __shared__ float hbuf[4 * 64];
  const int tid = threadIdx.x;
  for (int u = tid; u < 4096; u += 256) W2l[u] = W2[u];
  for (int u = tid; u < 384;  u += 256) W1l[u] = W1[u];
  __syncthreads();
  const int sub = tid >> 6, c = tid & 63;
  const int p = blockIdx.x * 4 + sub;
  const float b1c = b1[c], sc = bns[c], bb = bnb[c], b2c = b2[c];
  const float xi0 = pos[p * 3 + 0], xi1 = pos[p * 3 + 1], xi2 = pos[p * 3 + 2];
  float mx = -1e30f;
  for (int k = 0; k < KNN; ++k) {
    const int j = idx1[p * KNN + k];
    const float e3 = pos[j * 3 + 0] - xi0;
    const float e4 = pos[j * 3 + 1] - xi1;
    const float e5 = pos[j * 3 + 2] - xi2;
    float h = b1c + xi0 * W1l[0 * 64 + c] + xi1 * W1l[1 * 64 + c] + xi2 * W1l[2 * 64 + c]
                  + e3  * W1l[3 * 64 + c] + e4  * W1l[4 * 64 + c] + e5  * W1l[5 * 64 + c];
    h = fmaxf(h * sc + bb, 0.0f);
    hbuf[sub * 64 + c] = h;
    __syncthreads();
    float z = b2c;
#pragma unroll
    for (int r = 0; r < 64; ++r) z += hbuf[sub * 64 + r] * W2l[r * 64 + c];
    mx = fmaxf(mx, z);
    __syncthreads();
  }
  x1f[p * 64 + c]   = mx;
  xcat[p * 192 + c] = (_Float16)mx;           // concat cols 0..63
}

// ---------------------------------------------------------------------------
// Kernel 3: kNN in 64-D feature space. grid=128, block=256.
// ---------------------------------------------------------------------------
__global__ void knn64_kernel(const float* __restrict__ x1f, int* __restrict__ idx2) {
  __shared__ float tile[64 * 64];
  const int blk = blockIdx.x;
  const int b   = blk >> 2;
  const int i   = b * PP + (blk & 3) * 256 + threadIdx.x;
  float myx[64];
#pragma unroll
  for (int r = 0; r < 64; ++r) myx[r] = x1f[i * 64 + r];
  float bd[KNN]; int bi[KNN];
#pragma unroll
  for (int s = 0; s < KNN; ++s) { bd[s] = 1e30f; bi[s] = 0; }
  for (int t = 0; t < 16; ++t) {
    const int j0 = b * PP + t * 64;
    for (int u = threadIdx.x; u < 4096; u += 256) tile[u] = x1f[j0 * 64 + u];
    __syncthreads();
    for (int jj = 0; jj < 64; ++jj) {
      float d = 0.f;
#pragma unroll
      for (int r = 0; r < 64; ++r) {
        float df = myx[r] - tile[jj * 64 + r];
        d = fmaf(df, df, d);
      }
      topk_insert(bd, bi, d, j0 + jj);
    }
    __syncthreads();
  }
#pragma unroll
  for (int s = 0; s < KNN; ++s) idx2[i * KNN + s] = bi[s];
}

// ---------------------------------------------------------------------------
// B-fragment swizzled weight fill:
//   Bswz[((ks*8 + ct)*32 + lane)*16 + j] =
//        W[(ks*32 + 16*(j>=8) + 8*(lane>=16) + (j&7)) * 128 + ct*16 + (lane&15)]
// so each lane's 16-half fragment is contiguous (two ds_load_b128).
// total = KSTEPS * 8 * 32 * 16 halves. W is f32 row-major [K][128].
// ---------------------------------------------------------------------------
__device__ __forceinline__ void fill_bswz(_Float16* __restrict__ dst,
                                          const float* __restrict__ W,
                                          int total, int tid) {
  for (int u = tid; u < total; u += 256) {
    const int j    = u & 15;
    const int ln   = (u >> 4) & 31;
    const int ct   = (u >> 9) & 7;
    const int ks   = u >> 12;
    const int krow = ks * 32 + ((j >> 3) << 4) + ((ln >> 4) << 3) + (j & 7);
    const int col  = ct * 16 + (ln & 15);
    dst[u] = (_Float16)W[krow * 128 + col];
  }
}

// ---------------------------------------------------------------------------
// Kernel 4: EdgeConv2 via WMMA. One block (256 thr / 8 waves) handles 4 points.
// Per point: A = 32x128 f16 edge matrix (20 valid rows, rest zero) in LDS,
// B = conv2_W swizzled f16 in LDS. Wave w owns output col-tile w (16 cols),
// computes both 16-row tiles over 4 k-steps, then masked max over rows 0..19.
// ---------------------------------------------------------------------------
__global__ void edgeconv2_kernel(const float* __restrict__ x1f,
                                 const int*   __restrict__ idx2,
                                 const float* __restrict__ W,
                                 const float* __restrict__ bias,
                                 _Float16* __restrict__ xcat) {
  __shared__ _Float16 Wl[128 * 128];         // swizzled B fragments
  __shared__ _Float16 Abuf[32 * 128];
  const int tid  = threadIdx.x;
  fill_bswz(Wl, W, 128 * 128, tid);
  const int lane = tid & 31;
  const int wv   = tid >> 5;                 // col-tile 0..7
  const int l15  = lane & 15;
  const int hi8  = (lane >= 16) ? 8 : 0;

  for (int g = 0; g < 4; ++g) {
    const int p = blockIdx.x * 4 + g;
    __syncthreads();                          // Wl ready / prev Abuf consumed
    { // build A: rows = edges k=0..19, cols [x_i | x_j - x_i], pad rows w/ 0
      const int row = tid >> 3;               // 0..31
      const int c0  = (tid & 7) * 16;
      if (row < KNN) {
        const int j = idx2[p * KNN + row];
#pragma unroll
        for (int c = 0; c < 16; ++c) {
          const int col = c0 + c;
          float v;
          if (col < 64) v = x1f[p * 64 + col];
          else          v = x1f[j * 64 + (col - 64)] - x1f[p * 64 + (col - 64)];
          Abuf[row * 128 + col] = (_Float16)v;
        }
      } else {
#pragma unroll
        for (int c = 0; c < 16; ++c) Abuf[row * 128 + c0 + c] = (_Float16)0.0f;
      }
    }
    __syncthreads();

    v8f c0acc = {}; v8f c1acc = {};
#pragma unroll
    for (int ks = 0; ks < 4; ++ks) {
      const int kk = ks * 32;
      v8h a0l = *(const v8h*)&Abuf[l15 * 128 + kk + hi8];
      v8h a0h = *(const v8h*)&Abuf[l15 * 128 + kk + 16 + hi8];
      v8h a1l = *(const v8h*)&Abuf[(16 + l15) * 128 + kk + hi8];
      v8h a1h = *(const v8h*)&Abuf[(16 + l15) * 128 + kk + 16 + hi8];
      v16h a0 = __builtin_shufflevector(a0l, a0h, 0,1,2,3,4,5,6,7,8,9,10,11,12,13,14,15);
      v16h a1 = __builtin_shufflevector(a1l, a1h, 0,1,2,3,4,5,6,7,8,9,10,11,12,13,14,15);
      const _Float16* bp = &Wl[(((ks << 3) + wv) * 32 + lane) * 16];
      v8h blo = *(const v8h*)bp;
      v8h bhi = *(const v8h*)(bp + 8);
      v16h bf = __builtin_shufflevector(blo, bhi, 0,1,2,3,4,5,6,7,8,9,10,11,12,13,14,15);
      c0acc = __builtin_amdgcn_wmma_f32_16x16x32_f16(false, a0, false, bf, (short)0, c0acc, false, false);
      c1acc = __builtin_amdgcn_wmma_f32_16x16x32_f16(false, a1, false, bf, (short)0, c1acc, false, false);
    }
    // masked max over valid rows 0..19
    float m = -1e30f;
#pragma unroll
    for (int r = 0; r < 8; ++r) m = fmaxf(m, c0acc[r]);      // rows 0..15
    if (lane < 16) {
#pragma unroll
      for (int r = 0; r < 4; ++r) m = fmaxf(m, c1acc[r]);    // rows 16..19
    }
    float o = __shfl_xor(m, 16);
    m = fmaxf(m, o);
    if (lane < 16) {
      const int col = wv * 16 + lane;
      xcat[p * 192 + 64 + col] = (_Float16)(m + bias[col]);  // concat cols 64..191
    }
  }
}

// ---------------------------------------------------------------------------
// Kernel 5: lin GEMM via WMMA: [32768,192] x [192,128] + bias -> h f32.
// grid=512, block=256 (8 waves). Block covers 64 rows; wave w: row-tile w/2,
// col-tiles (w&1)*4 .. +3. A frags straight from global f16, B swizzled in LDS.
// ---------------------------------------------------------------------------
__global__ void lin_kernel(const _Float16* __restrict__ xcat,
                           const float* __restrict__ W,
                           const float* __restrict__ bias,
                           float* __restrict__ hout) {
  __shared__ _Float16 Wl[192 * 128];         // swizzled B fragments (6 k-steps)
  const int tid = threadIdx.x;
  fill_bswz(Wl, W, 192 * 128, tid);
  __syncthreads();
  const int lane = tid & 31;
  const int wv   = tid >> 5;
  const int rt   = wv >> 1;
  const int cg   = (wv & 1) * 4;
  const int l15  = lane & 15;
  const int hi8  = (lane >= 16) ? 8 : 0;
  const int rbase = blockIdx.x * 64;
  const int arow  = rbase + rt * 16 + l15;
  v8f acc[4];
#pragma unroll
  for (int t = 0; t < 4; ++t) acc[t] = (v8f){};
#pragma unroll
  for (int ks = 0; ks < 6; ++ks) {
    const int kk = ks * 32;
    v8h alo = *(const v8h*)&xcat[arow * 192 + kk + hi8];
    v8h ahi = *(const v8h*)&xcat[arow * 192 + kk + 16 + hi8];
    v16h a = __builtin_shufflevector(alo, ahi, 0,1,2,3,4,5,6,7,8,9,10,11,12,13,14,15);
#pragma unroll
    for (int ct = 0; ct < 4; ++ct) {
      const _Float16* bp = &Wl[(((ks << 3) + (cg + ct)) * 32 + lane) * 16];
      v8h blo = *(const v8h*)bp;
      v8h bhi = *(const v8h*)(bp + 8);
      v16h bf = __builtin_shufflevector(blo, bhi, 0,1,2,3,4,5,6,7,8,9,10,11,12,13,14,15);
      acc[ct] = __builtin_amdgcn_wmma_f32_16x16x32_f16(false, a, false, bf, (short)0, acc[ct], false, false);
    }
  }
#pragma unroll
  for (int ct = 0; ct < 4; ++ct) {
    const int col = (cg + ct) * 16 + l15;
    const float bcol = bias[col];
#pragma unroll
    for (int r = 0; r < 8; ++r) {
      const int row = rbase + rt * 16 + r + hi8;
      hout[row * 128 + col] = acc[ct][r] + bcol;
    }
  }
}

// ---------------------------------------------------------------------------
// Kernel 6: global max pool per cloud. grid=32, block=128.
// ---------------------------------------------------------------------------
__global__ void segmax_kernel(const float* __restrict__ h, float* __restrict__ out) {
  const int b = blockIdx.x, c = threadIdx.x;
  float m = -1e30f;
  for (int p = 0; p < PP; ++p) m = fmaxf(m, h[(b * PP + p) * 128 + c]);
  out[b * 128 + c] = m;
}

// ---------------------------------------------------------------------------
// workspace layout (bytes)
// ---------------------------------------------------------------------------
static constexpr size_t OFF_IDX1 = 0;                                   // N*K int
static constexpr size_t OFF_IDX2 = OFF_IDX1 + (size_t)NTOT * KNN * 4;   // N*K int
static constexpr size_t OFF_X1F  = OFF_IDX2 + (size_t)NTOT * KNN * 4;   // N*64 f32
static constexpr size_t OFF_XCAT = OFF_X1F  + (size_t)NTOT * 64 * 4;    // N*192 f16
static constexpr size_t OFF_H    = OFF_XCAT + (size_t)NTOT * 192 * 2;   // N*128 f32

extern "C" void kernel_launch(void* const* d_in, const int* in_sizes, int n_in,
                              void* d_out, int out_size, void* d_ws, size_t ws_size,
                              hipStream_t stream) {
  (void)in_sizes; (void)n_in; (void)out_size; (void)ws_size;
  const float* pos   = (const float*)d_in[0];
  // d_in[1] = batch ids (clouds are contiguous; index arithmetic replaces it)
  const float* c1W1  = (const float*)d_in[2];
  const float* c1b1  = (const float*)d_in[3];
  const float* c1bns = (const float*)d_in[4];
  const float* c1bnb = (const float*)d_in[5];
  const float* c1W2  = (const float*)d_in[6];
  const float* c1b2  = (const float*)d_in[7];
  const float* c2W   = (const float*)d_in[8];
  const float* c2b   = (const float*)d_in[9];
  const float* linW  = (const float*)d_in[10];
  const float* linb  = (const float*)d_in[11];

  char* ws = (char*)d_ws;
  int*      idx1 = (int*)(ws + OFF_IDX1);
  int*      idx2 = (int*)(ws + OFF_IDX2);
  float*    x1f  = (float*)(ws + OFF_X1F);
  _Float16* xcat = (_Float16*)(ws + OFF_XCAT);
  float*    hbuf = (float*)(ws + OFF_H);

  knn3_kernel     <<<BN * 4,    256, 0, stream>>>(pos, idx1);
  edgeconv1_kernel<<<NTOT / 4,  256, 0, stream>>>(pos, idx1, c1W1, c1b1, c1bns, c1bnb,
                                                  c1W2, c1b2, x1f, xcat);
  knn64_kernel    <<<BN * 4,    256, 0, stream>>>(x1f, idx2);
  edgeconv2_kernel<<<NTOT / 4,  256, 0, stream>>>(x1f, idx2, c2W, c2b, xcat);
  lin_kernel      <<<NTOT / 64, 256, 0, stream>>>(xcat, linW, linb, hbuf);
  segmax_kernel   <<<BN,        128, 0, stream>>>(hbuf, (float*)d_out);
}
